// MambaMIL_43868795961659
// MI455X (gfx1250) — matrix operand order repaired and checked by reading, not compiled, not run on hardware
//
#include <hip/hip_runtime.h>
#include <cmath>

// ---------------------------------------------------------------------------
// MambaMIL forward for MI455X (gfx1250): bf16 WMMA GEMMs + wave32 scan.
// ---------------------------------------------------------------------------

typedef __attribute__((ext_vector_type(16))) __bf16 v16bf;
typedef __attribute__((ext_vector_type(8)))  __bf16 v8bf;
typedef __attribute__((ext_vector_type(8)))  float  v8f;

#define SEQ     8192
#define DMODEL  512
#define DINNER  1024
#define DSTATE  16
#define DTRANK  32

enum { ACT_NONE = 0, ACT_GELU = 1, ACT_SOFTPLUS = 2, ACT_TANH = 3 };

__device__ __forceinline__ float wave_sum32(float v) {
  v += __shfl_xor(v, 16, 32);
  v += __shfl_xor(v, 8, 32);
  v += __shfl_xor(v, 4, 32);
  v += __shfl_xor(v, 2, 32);
  v += __shfl_xor(v, 1, 32);
  return v;
}

// ---------------------------------------------------------------------------
// Tiled WMMA GEMM:  C[M,N] = act( A[M,K] @ W[N,K]^T + bias ) (+ res)
// Block = 256 threads (8 waves), tile 128x64, K-step 32.
// Wave (wm 0..3, wn 0..1) owns a 32x32 sub-block: 2 A-frags x 2 B-frags
// -> 4 WMMAs per K-step with full register reuse of both fragments.
// Staging issues all 6 global b128 loads before any convert/store so the
// backend can overlap them (LOADcnt 5..0), then one uniform prefetch branch.
// Requires: M%128==0, N%64==0, K%32==0 (true for every GEMM in this model).
// ---------------------------------------------------------------------------
template <int ACT>
__global__ __launch_bounds__(256) void gemm_bf16_wmma(
    const float* __restrict__ A, int lda,
    const float* __restrict__ W,
    const float* __restrict__ bias,
    const float* __restrict__ res,
    float* __restrict__ C, int N, int K) {
  __shared__ __bf16 As[128 * 32];
  __shared__ __bf16 Ws[64 * 32];

  const int tid  = threadIdx.x;
  const int lane = tid & 31, wid = tid >> 5;
  const int wm = wid & 3, wn = wid >> 2;
  const int hl = lane >> 4, r = lane & 15;
  const int m0 = blockIdx.y * 128, n0 = blockIdx.x * 64;

  v8f acc[2][2];
#pragma unroll
  for (int mi = 0; mi < 2; ++mi)
#pragma unroll
    for (int nj = 0; nj < 2; ++nj)
#pragma unroll
      for (int e = 0; e < 8; ++e) acc[mi][nj][e] = 0.f;

  for (int k0 = 0; k0 < K; k0 += 32) {
    __syncthreads();
    // ---- Stage A (128x32) and W (64x32) as bf16: batch all loads first ----
    float4 av[4], wv[2];
#pragma unroll
    for (int i = 0; i < 4; ++i) {
      int idx = i * 256 + tid;               // 0..1023 float4 slots, coalesced
      int row = idx >> 3, c4 = (idx & 7) * 4;
      av[i] = *(const float4*)(A + (size_t)(m0 + row) * lda + k0 + c4);
    }
#pragma unroll
    for (int i = 0; i < 2; ++i) {
      int idx = i * 256 + tid;               // 0..511 float4 slots
      int row = idx >> 3, c4 = (idx & 7) * 4;
      wv[i] = *(const float4*)(W + (size_t)(n0 + row) * K + k0 + c4);
    }
    // One uniform branch: threads 0..127 prefetch A rows, 128..191 W rows.
    if (k0 + 32 < K) {                       // gfx1250 global_prefetch_b8
      if (tid < 128)
        __builtin_prefetch(A + (size_t)(m0 + tid) * lda + k0 + 32, 0, 1);
      else if (tid < 192)
        __builtin_prefetch(W + (size_t)(n0 + tid - 128) * K + k0 + 32, 0, 1);
    }
    // ---- Convert and store to LDS ----
#pragma unroll
    for (int i = 0; i < 4; ++i) {
      int idx = i * 256 + tid;
      int row = idx >> 3, c4 = (idx & 7) * 4;
      __bf16* ad = &As[row * 32 + c4];
      ad[0] = (__bf16)av[i].x; ad[1] = (__bf16)av[i].y;
      ad[2] = (__bf16)av[i].z; ad[3] = (__bf16)av[i].w;
    }
#pragma unroll
    for (int i = 0; i < 2; ++i) {
      int idx = i * 256 + tid;
      int row = idx >> 3, c4 = (idx & 7) * 4;
      __bf16* wd = &Ws[row * 32 + c4];
      wd[0] = (__bf16)wv[i].x; wd[1] = (__bf16)wv[i].y;
      wd[2] = (__bf16)wv[i].z; wd[3] = (__bf16)wv[i].w;
    }
    __syncthreads();

    // A fragments: lane (hl,r) -> row r, K chunks [hl*8 ..] and [16+hl*8 ..]
    v16bf af[2];
#pragma unroll
    for (int mi = 0; mi < 2; ++mi) {
      const __bf16* ap = &As[(wm * 32 + mi * 16 + r) * 32 + hl * 8];
      v8bf a0 = *(const v8bf*)ap;
      v8bf a1 = *(const v8bf*)(ap + 16);
#pragma unroll
      for (int e = 0; e < 8; ++e) { af[mi][e] = a0[e]; af[mi][8 + e] = a1[e]; }
    }
    // B fragments: lane (hl,r) -> column r, K chunks [hl*8 ..] and [16+hl*8 ..]
    v16bf bf[2];
#pragma unroll
    for (int nj = 0; nj < 2; ++nj) {
      const __bf16* bp = &Ws[(wn * 32 + nj * 16 + r) * 32 + hl * 8];
      v8bf b0 = *(const v8bf*)bp;
      v8bf b1 = *(const v8bf*)(bp + 16);
#pragma unroll
      for (int e = 0; e < 8; ++e) { bf[nj][e] = b0[e]; bf[nj][8 + e] = b1[e]; }
    }

#pragma unroll
    for (int mi = 0; mi < 2; ++mi)
#pragma unroll
      for (int nj = 0; nj < 2; ++nj)
        acc[mi][nj] = __builtin_amdgcn_wmma_f32_16x16x32_bf16(
            false, af[mi], false, bf[nj], (short)0, acc[mi][nj], false, false);
  }

  // Epilogue: D layout -> lane (hl,r), vgpr rr = element (M=hl*8+rr, N=r).
#pragma unroll
  for (int mi = 0; mi < 2; ++mi) {
    const int mbase = m0 + wm * 32 + mi * 16 + hl * 8;
#pragma unroll
    for (int nj = 0; nj < 2; ++nj) {
      int n = n0 + wn * 32 + nj * 16 + r;
      float bv = bias ? bias[n] : 0.f;
#pragma unroll
      for (int rr = 0; rr < 8; ++rr) {
        float v = acc[mi][nj][rr] + bv;
        if (ACT == ACT_GELU)          v = 0.5f * v * (1.f + erff(v * 0.70710678118654752f));
        else if (ACT == ACT_SOFTPLUS) v = (v > 20.f) ? v : log1pf(__expf(v));
        else if (ACT == ACT_TANH)     v = tanhf(v);
        size_t oidx = (size_t)(mbase + rr) * N + n;
        if (res) v += res[oidx];
        C[oidx] = v;
      }
    }
  }
}

// ---------------------------------------------------------------------------
// LayerNorm over 512 features, one wave per row (16 elems/lane, shfl reduce).
// ---------------------------------------------------------------------------
__global__ __launch_bounds__(256) void layernorm_kernel(
    const float* __restrict__ x, const float* __restrict__ w,
    const float* __restrict__ b, float* __restrict__ out) {
  int wid = threadIdx.x >> 5, lane = threadIdx.x & 31;
  size_t row = (size_t)blockIdx.x * 8 + wid;
  const float* xr = x + row * DMODEL;
  float v[16];
  float s = 0.f;
#pragma unroll
  for (int i = 0; i < 4; ++i) {
    float4 t = *(const float4*)(xr + i * 128 + lane * 4);
    v[i * 4 + 0] = t.x; v[i * 4 + 1] = t.y; v[i * 4 + 2] = t.z; v[i * 4 + 3] = t.w;
    s += t.x + t.y + t.z + t.w;
  }
  s = wave_sum32(s);
  float mu = s * (1.f / 512.f);
  float var = 0.f;
#pragma unroll
  for (int k = 0; k < 16; ++k) { float d0 = v[k] - mu; var += d0 * d0; }
  var = wave_sum32(var);
  float inv = rsqrtf(var * (1.f / 512.f) + 1e-5f);
  float* orow = out + row * DMODEL;
#pragma unroll
  for (int i = 0; i < 4; ++i) {
    int cb = i * 128 + lane * 4;
    float4 wv = *(const float4*)(w + cb);
    float4 bv = *(const float4*)(b + cb);
    float4 o;
    o.x = (v[i * 4 + 0] - mu) * inv * wv.x + bv.x;
    o.y = (v[i * 4 + 1] - mu) * inv * wv.y + bv.y;
    o.z = (v[i * 4 + 2] - mu) * inv * wv.z + bv.z;
    o.w = (v[i * 4 + 3] - mu) * inv * wv.w + bv.w;
    *(float4*)(orow + cb) = o;
  }
}

// ---------------------------------------------------------------------------
// Causal depthwise conv (k=4) over u = xz[:, :1024], + bias, + SiLU.
// ---------------------------------------------------------------------------
__global__ __launch_bounds__(256) void conv_silu_kernel(
    const float* __restrict__ xz, const float* __restrict__ w,
    const float* __restrict__ b, float* __restrict__ out) {
  int idx = blockIdx.x * 256 + threadIdx.x;   // t*1024 + d
  int d = idx & (DINNER - 1);
  int t = idx >> 10;
  float4 wv = *(const float4*)(w + d * 4);
  const float* col = xz + d;
  float acc = b[d];
  if (t >= 3) acc += wv.x * col[(size_t)(t - 3) * 2048];
  if (t >= 2) acc += wv.y * col[(size_t)(t - 2) * 2048];
  if (t >= 1) acc += wv.z * col[(size_t)(t - 1) * 2048];
  acc += wv.w * col[(size_t)t * 2048];
  out[idx] = acc / (1.f + __expf(-acc));      // silu
}

// ---------------------------------------------------------------------------
// Selective scan. One lane per (channel d, state n): 16 lanes/channel,
// 16 channels per 256-thread block -> grid 64, 512 waves total.
// Recurrence kept in registers; y reduced with a 16-lane shfl butterfly.
// Fused epilogue: y = (y + u*Dp) * silu(z).
// ---------------------------------------------------------------------------
__global__ __launch_bounds__(256) void scan_kernel(
    const float* __restrict__ uc, const float* __restrict__ dt,
    const float* __restrict__ proj, const float* __restrict__ A_log,
    const float* __restrict__ Dp, const float* __restrict__ xz,
    float* __restrict__ y) {
  int n = threadIdx.x & 15;
  int d = blockIdx.x * 16 + (threadIdx.x >> 4);
  float A  = -__expf(A_log[(size_t)d * DSTATE + n]);
  float dp = Dp[d];
  float hs = 0.f;
  for (int t = 0; t < SEQ; ++t) {
    float dtv = dt[(size_t)t * DINNER + d];
    float uv  = uc[(size_t)t * DINNER + d];
    float bn  = proj[(size_t)t * 64 + 32 + n];
    float cn  = proj[(size_t)t * 64 + 48 + n];
    hs = __expf(dtv * A) * hs + (dtv * uv) * bn;
    float p = hs * cn;
    p += __shfl_xor(p, 1, 32);
    p += __shfl_xor(p, 2, 32);
    p += __shfl_xor(p, 4, 32);
    p += __shfl_xor(p, 8, 32);
    if (n == 0) {
      float zv = xz[(size_t)t * 2048 + DINNER + d];
      float yv = p + uv * dp;
      y[(size_t)t * DINNER + d] = yv * (zv / (1.f + __expf(-zv)));
    }
  }
}

// ---------------------------------------------------------------------------
// Attention pooling head.
// ---------------------------------------------------------------------------
__global__ __launch_bounds__(256) void attn_score_kernel(
    const float* __restrict__ tb, const float* __restrict__ w2,
    const float* __restrict__ b2, float* __restrict__ s) {
  int wid = threadIdx.x >> 5, lane = threadIdx.x & 31;
  size_t row = (size_t)blockIdx.x * 8 + wid;
  float4 a = *(const float4*)(tb + row * 128 + lane * 4);
  float4 w = *(const float4*)(w2 + lane * 4);
  float p = a.x * w.x + a.y * w.y + a.z * w.z + a.w * w.w;
  p = wave_sum32(p);
  if (lane == 0) s[row] = p + b2[0];
}

__global__ __launch_bounds__(1024) void softmax_max_kernel(
    const float* __restrict__ s, float* __restrict__ smax) {
  __shared__ float red[32];
  int tid = threadIdx.x, lane = tid & 31, wid = tid >> 5;
  float m = -3.4e38f;
  for (int i = tid; i < SEQ; i += 1024) m = fmaxf(m, s[i]);
#pragma unroll
  for (int o = 16; o > 0; o >>= 1) m = fmaxf(m, __shfl_xor(m, o, 32));
  if (lane == 0) red[wid] = m;
  __syncthreads();
  if (tid < 32) {
    float mm = red[tid];
#pragma unroll
    for (int o = 16; o > 0; o >>= 1) mm = fmaxf(mm, __shfl_xor(mm, o, 32));
    if (tid == 0) smax[0] = mm;
  }
}

__global__ __launch_bounds__(1024) void softmax_expsum_kernel(
    const float* __restrict__ s, const float* __restrict__ smax,
    float* __restrict__ e, float* __restrict__ ssum) {
  __shared__ float red[32];
  int tid = threadIdx.x, lane = tid & 31, wid = tid >> 5;
  float mx = smax[0];
  float acc = 0.f;
  for (int i = tid; i < SEQ; i += 1024) {
    float ev = __expf(s[i] - mx);
    e[i] = ev;
    acc += ev;
  }
  acc = wave_sum32(acc);
  if (lane == 0) red[wid] = acc;
  __syncthreads();
  if (tid < 32) {
    float a = red[tid];
    a = wave_sum32(a);
    if (tid == 0) ssum[0] = a;
  }
}

__global__ __launch_bounds__(256) void pooled_kernel(
    const float* __restrict__ e, const float* __restrict__ hn,
    const float* __restrict__ ssum, float* __restrict__ pooled) {
  __shared__ float red[8];
  int c = blockIdx.x, tid = threadIdx.x, lane = tid & 31, wid = tid >> 5;
  float acc = 0.f;
  for (int t = tid; t < SEQ; t += 256) acc += e[t] * hn[(size_t)t * DMODEL + c];
  acc = wave_sum32(acc);
  if (lane == 0) red[wid] = acc;
  __syncthreads();
  if (tid == 0) {
    float a = 0.f;
#pragma unroll
    for (int i = 0; i < 8; ++i) a += red[i];
    pooled[c] = a / ssum[0];
  }
}

__global__ void logits_kernel(const float* __restrict__ pooled,
                              const float* __restrict__ cw,
                              const float* __restrict__ cb,
                              float* __restrict__ out) {
  int wid = threadIdx.x >> 5, lane = threadIdx.x & 31;  // 2 waves, one per class
  float acc = 0.f;
  for (int i = lane; i < DMODEL; i += 32) acc += pooled[i] * cw[wid * DMODEL + i];
  acc = wave_sum32(acc);
  if (lane == 0) out[wid] = acc + cb[wid];
}

// ---------------------------------------------------------------------------
extern "C" void kernel_launch(void* const* d_in, const int* in_sizes, int n_in,
                              void* d_out, int out_size, void* d_ws, size_t ws_size,
                              hipStream_t stream) {
  const float* x         = (const float*)d_in[0];
  const float* fc1_w     = (const float*)d_in[1];
  const float* fc1_b     = (const float*)d_in[2];
  const float* ln_w      = (const float*)d_in[3];
  const float* ln_b      = (const float*)d_in[4];
  const float* in_proj_w = (const float*)d_in[5];
  const float* conv_w    = (const float*)d_in[6];
  const float* conv_b    = (const float*)d_in[7];
  const float* x_proj_w  = (const float*)d_in[8];
  const float* dt_proj_w = (const float*)d_in[9];
  const float* dt_proj_b = (const float*)d_in[10];
  const float* A_log     = (const float*)d_in[11];
  const float* D_param   = (const float*)d_in[12];
  const float* out_proj_w= (const float*)d_in[13];
  const float* norm_w    = (const float*)d_in[14];
  const float* norm_b    = (const float*)d_in[15];
  const float* attn1_w   = (const float*)d_in[16];
  const float* attn1_b   = (const float*)d_in[17];
  const float* attn2_w   = (const float*)d_in[18];
  const float* attn2_b   = (const float*)d_in[19];
  const float* clf_w     = (const float*)d_in[20];
  const float* clf_b     = (const float*)d_in[21];

  float* ws = (float*)d_ws;
  size_t off = 0;
  float* h      = ws + off; off += (size_t)SEQ * DMODEL;     // 4M
  float* hn     = ws + off; off += (size_t)SEQ * DMODEL;     // 4M
  float* xz     = ws + off; off += (size_t)SEQ * 2 * DINNER; // 16M
  float* uc     = ws + off; off += (size_t)SEQ * DINNER;     // 8M
  float* proj   = ws + off; off += (size_t)SEQ * 64;         // 0.5M
  float* dtb    = ws + off; off += (size_t)SEQ * DINNER;     // 8M
  float* yb     = ws + off; off += (size_t)SEQ * DINNER;     // 8M
  float* tb     = ws + off; off += (size_t)SEQ * 128;        // 1M
  float* sb     = ws + off; off += SEQ;
  float* eb     = ws + off; off += SEQ;
  float* smax   = ws + off; off += 1;
  float* ssum   = ws + off; off += 1;
  float* pooled = ws + off; off += DMODEL;
  (void)in_sizes; (void)n_in; (void)out_size; (void)ws_size;

  const dim3 B(256);
  const int MT = SEQ / 128;  // 64 M-tiles

  // h = gelu(x @ fc1_w^T + fc1_b)
  gemm_bf16_wmma<ACT_GELU><<<dim3(DMODEL / 64, MT), B, 0, stream>>>(
      x, 1024, fc1_w, fc1_b, nullptr, h, DMODEL, 1024);

  for (int L = 0; L < 2; ++L) {
    layernorm_kernel<<<SEQ / 8, B, 0, stream>>>(h, ln_w + L * DMODEL,
                                                ln_b + L * DMODEL, hn);
    // xz = hn @ in_proj_w^T
    gemm_bf16_wmma<ACT_NONE><<<dim3(2 * DINNER / 64, MT), B, 0, stream>>>(
        hn, DMODEL, in_proj_w + (size_t)L * 2 * DINNER * DMODEL,
        nullptr, nullptr, xz, 2 * DINNER, DMODEL);
    // uc = silu(depthwise_conv(xz[:, :1024]) + conv_b)
    conv_silu_kernel<<<(SEQ * DINNER) / 256, B, 0, stream>>>(
        xz, conv_w + (size_t)L * DINNER * 4, conv_b + L * DINNER, uc);
    // proj = uc @ x_proj_w^T   (cols 0..31 dt-rank, 32..47 B, 48..63 C)
    gemm_bf16_wmma<ACT_NONE><<<dim3(1, MT), B, 0, stream>>>(
        uc, DINNER, x_proj_w + (size_t)L * 64 * DINNER,
        nullptr, nullptr, proj, 64, DINNER);
    // dtb = softplus(proj[:, :32] @ dt_proj_w^T + dt_proj_b)
    gemm_bf16_wmma<ACT_SOFTPLUS><<<dim3(DINNER / 64, MT), B, 0, stream>>>(
        proj, 64, dt_proj_w + (size_t)L * DINNER * DTRANK,
        dt_proj_b + L * DINNER, nullptr, dtb, DINNER, DTRANK);
    // yb = selective_scan(...) * silu(z)
    scan_kernel<<<DINNER / 16, B, 0, stream>>>(
        uc, dtb, proj, A_log + (size_t)L * DINNER * DSTATE,
        D_param + L * DINNER, xz, yb);
    // h = yb @ out_proj_w^T + h   (residual, in place)
    gemm_bf16_wmma<ACT_NONE><<<dim3(DMODEL / 64, MT), B, 0, stream>>>(
        yb, DINNER, out_proj_w + (size_t)L * DMODEL * DINNER,
        nullptr, h, h, DMODEL, DINNER);
  }

  // hn = layer_norm(h); attention pooling + classifier
  layernorm_kernel<<<SEQ / 8, B, 0, stream>>>(h, norm_w, norm_b, hn);
  gemm_bf16_wmma<ACT_TANH><<<dim3(2, MT), B, 0, stream>>>(
      hn, DMODEL, attn1_w, attn1_b, nullptr, tb, 128, DMODEL);
  attn_score_kernel<<<SEQ / 8, B, 0, stream>>>(tb, attn2_w, attn2_b, sb);
  softmax_max_kernel<<<1, 1024, 0, stream>>>(sb, smax);
  softmax_expsum_kernel<<<1, 1024, 0, stream>>>(sb, smax, eb, ssum);
  pooled_kernel<<<DMODEL, B, 0, stream>>>(eb, hn, ssum, pooled);
  logits_kernel<<<1, 64, 0, stream>>>(pooled, clf_w, clf_b, (float*)d_out);
}
